// Net_51187420234578
// MI455X (gfx1250) — matrix operand order repaired and checked by reading, not compiled, not run on hardware
//
#include <hip/hip_runtime.h>
#include <hip/hip_bf16.h>

typedef __attribute__((ext_vector_type(16))) _Float16 v16h;
typedef __attribute__((ext_vector_type(8)))  float    v8f;

#define IMGD 401
#define HW   (401*401)          // 160801
#define NB   2
#define NPIX (NB*HW)            // 321602
#define CIN  2048
#define CTOT 150                // 17 kp + 34 so + 64 mo + 34 lo + 1 ss
#define MROWS 338               // 2*13*13

// directed-edge -> target keypoint (EDGES[i][1] then EDGES[i][0])
__constant__ int TO_KP[32] = {
    1,3,2,4,5,7,9,11,13,15,6,8,10,12,14,16,
    0,1,0,2,0,5,7,5,11,13,0,6,8,6,12,14
};

// ---------------------------------------------------------------------------
// Kernel 1: [338 x 2048] x [2048 x 150] GEMM via WMMA f32<-f16, fused bias +
// sigmoid (kp cols 0..16, ss col 149). One wave per 16x16 output tile.
// Out-of-range rows/cols are CLAMPED on the load side (their D entries are
// never stored), so the K-loop is branch-free with full EXEC throughout.
// ---------------------------------------------------------------------------
__device__ inline void col_map(int c,
                               const float* kp_w, const float* so_w, const float* mo_w,
                               const float* lo_w, const float* ss_w,
                               const float* kp_b, const float* so_b, const float* mo_b,
                               const float* lo_b, const float* ss_b,
                               const float*& w, int& width, int& col, float& bias, bool& sig) {
    if (c < 17)        { w = kp_w; width = 17; col = c;       bias = kp_b[col]; sig = true;  }
    else if (c < 51)   { w = so_w; width = 34; col = c - 17;  bias = so_b[col]; sig = false; }
    else if (c < 115)  { w = mo_w; width = 64; col = c - 51;  bias = mo_b[col]; sig = false; }
    else if (c < 149)  { w = lo_w; width = 34; col = c - 115; bias = lo_b[col]; sig = false; }
    else               { w = ss_w; width = 1;  col = 0;       bias = ss_b[0];   sig = true;  }
}

__global__ __launch_bounds__(32)
void gemm150_wmma(const float* __restrict__ x,
                  const float* __restrict__ kp_w, const float* __restrict__ kp_b,
                  const float* __restrict__ so_w, const float* __restrict__ so_b,
                  const float* __restrict__ mo_w, const float* __restrict__ mo_b,
                  const float* __restrict__ lo_w, const float* __restrict__ lo_b,
                  const float* __restrict__ ss_w, const float* __restrict__ ss_b,
                  float* __restrict__ out /* [338][150] */) {
    const int mtile = blockIdx.x;            // 0..21  (22*16 = 352 >= 338)
    const int ntile = blockIdx.y;            // 0..9   (10*16 = 160 >= 150)
    const int lane  = threadIdx.x;           // wave32

    // A as loaded (contiguous k, as B): lane<16 -> M=lane, K base 0;
    // lane>=16 -> M=lane-16, K base 16.  a[i]: K = kbase + i.
    const int arow   = min(mtile * 16 + (lane & 15), MROWS - 1);   // clamped
    const int kbaseA = (lane < 16) ? 0 : 16;
    // B as loaded (contiguous k, as A): lanes 0-15 hold K=0..15 (N=lane), lanes 16-31 K=16..31.
    const int ncol   = ntile * 16 + (lane & 15);
    const int kbaseB = (lane < 16) ? 0 : 16;

    const float* wsrc; int width, wcol; float bias; bool sig;
    col_map(ncol, kp_w, so_w, mo_w, lo_w, ss_w, kp_b, so_b, mo_b, lo_b, ss_b,
            wsrc, width, wcol, bias, sig);

    // Running pointers: one 64-bit add per operand per K-step.
    const float* ap = x + (size_t)arow * CIN + kbaseA;
    const float* bp = wsrc + (size_t)kbaseB * width + wcol;
    const size_t bstep = (size_t)32 * width;

    v8f acc = {};
    for (int k0 = 0; k0 < CIN; k0 += 32) {
        v16h a, b;
#pragma unroll
        for (int i = 0; i < 8; ++i)  a[i] = (_Float16)ap[i];
#pragma unroll
        for (int i = 8; i < 16; ++i) a[i] = (_Float16)ap[i];
#pragma unroll
        for (int i = 0; i < 16; ++i) b[i] = (_Float16)bp[(size_t)i * width];
        // hint the next K-slab into cache (global_prefetch_b8)
        __builtin_prefetch(ap + 32, 0, 0);
        __builtin_prefetch(bp + bstep, 0, 0);
        acc = __builtin_amdgcn_wmma_f32_16x16x32_f16(
            /*neg_a=*/false, a, /*neg_b=*/false, b,
            /*c_mod=*/(short)0, acc, /*reuse_a=*/false, /*reuse_b=*/false);
        ap += 32;
        bp += bstep;
    }

    // C/D layout: VGPR r -> M = mtile*16 + r (+8 for lanes 16-31), N = lane&15.
    const int mofs = (lane < 16) ? 0 : 8;
#pragma unroll
    for (int r = 0; r < 8; ++r) {
        const int row = mtile * 16 + r + mofs;
        if (row < MROWS && ncol < CTOT) {
            float v = acc[r] + bias;
            if (sig) v = 1.0f / (1.0f + __expf(-v));
            out[(size_t)row * CTOT + ncol] = v;
        }
    }
}

// ---------------------------------------------------------------------------
// Bilinear resize helpers (jax.image.resize 'bilinear': half-pixel centers,
// edge clamp), 13 -> 401.
// ---------------------------------------------------------------------------
__device__ inline void rs_axis(int d, int& i0, int& i1, float& w) {
    float src = ((float)d + 0.5f) * (13.0f / 401.0f) - 0.5f;
    float f = floorf(src);
    w = src - f;
    int i = (int)f;
    i0 = min(max(i, 0), 12);
    i1 = min(max(i + 1, 0), 12);
}

// ---------------------------------------------------------------------------
// Kernel 2: resize 13x13x150 -> 401x401; write kp/so/ss to d_out, L (lo) to ws.
// ---------------------------------------------------------------------------
__global__ __launch_bounds__(256)
void resize_scatter(const float* __restrict__ small,
                    float* __restrict__ out_kp, float* __restrict__ out_so,
                    float* __restrict__ out_ss, float* __restrict__ Lfield) {
    const int pix = blockIdx.x * blockDim.x + threadIdx.x;
    if (pix >= NPIX) return;
    const int n = pix / HW;
    const int p = pix % HW;
    const int y = p / IMGD, x = p % IMGD;

    int x0, x1, y0, y1; float wx, wy;
    rs_axis(x, x0, x1, wx);
    rs_axis(y, y0, y1, wy);
    const float w00 = (1.0f - wy) * (1.0f - wx);
    const float w01 = (1.0f - wy) * wx;
    const float w10 = wy * (1.0f - wx);
    const float w11 = wy * wx;
    const float* s00 = small + (size_t)((n * 13 + y0) * 13 + x0) * CTOT;
    const float* s01 = small + (size_t)((n * 13 + y0) * 13 + x1) * CTOT;
    const float* s10 = small + (size_t)((n * 13 + y1) * 13 + x0) * CTOT;
    const float* s11 = small + (size_t)((n * 13 + y1) * 13 + x1) * CTOT;

    float* okp = out_kp + (size_t)pix * 17;
#pragma unroll 17
    for (int c = 0; c < 17; ++c)
        okp[c] = w00 * s00[c] + w01 * s01[c] + w10 * s10[c] + w11 * s11[c];

    float* oso = out_so + (size_t)pix * 34;
#pragma unroll 17
    for (int c = 0; c < 34; ++c)
        oso[c] = w00 * s00[17 + c] + w01 * s01[17 + c] + w10 * s10[17 + c] + w11 * s11[17 + c];

    float* oL = Lfield + (size_t)pix * 34;
#pragma unroll 17
    for (int c = 0; c < 34; ++c)
        oL[c] = w00 * s00[115 + c] + w01 * s01[115 + c] + w10 * s10[115 + c] + w11 * s11[115 + c];

    out_ss[pix] = w00 * s00[149] + w01 * s01[149] + w10 * s10[149] + w11 * s11[149];
}

// ---------------------------------------------------------------------------
// Faithful port of the TF/JAX bilinear_sampler (pad-1 border, floor/ceil
// corners, OOB -> zero). Fn points at batch n of a [HW][34] field.
// ---------------------------------------------------------------------------
__device__ inline void sample34(const float* __restrict__ Fn, int ch, int x, int y,
                                float vx, float vy, float& ox, float& oy) {
    const float vx0 = floorf(vx), vy0 = floorf(vy);
    const float vx1 = ceilf(vx),  vy1 = ceilf(vy);
    const float fx0 = vx0 + (float)(x + 1);
    const float fx1 = vx1 + (float)(x + 1);
    const float fy0 = vy0 + (float)(y + 1);
    const float fy1 = vy1 + (float)(y + 1);
    if (fx0 < 1.0f || fy0 < 1.0f || fx1 > 401.0f || fy1 > 401.0f) {
        ox = 0.0f; oy = 0.0f; return;
    }
    const int X0 = (int)fx0 - 1, X1 = (int)fx1 - 1;
    const int Y0 = (int)fy0 - 1, Y1 = (int)fy1 - 1;
    const float dx = vx - vx0, dy = vy - vy0;
    const float w00 = (1.0f - dx) * (1.0f - dy);
    const float w01 = (1.0f - dx) * dy;
    const float w10 = dx * (1.0f - dy);
    const float w11 = dx * dy;
    const float* p00 = Fn + (size_t)(Y0 * IMGD + X0) * 34 + ch;
    const float* p01 = Fn + (size_t)(Y1 * IMGD + X0) * 34 + ch;
    const float* p10 = Fn + (size_t)(Y0 * IMGD + X1) * 34 + ch;
    const float* p11 = Fn + (size_t)(Y1 * IMGD + X1) * 34 + ch;
    ox = w00 * p00[0] + w01 * p01[0] + w10 * p10[0] + w11 * p11[0];
    oy = w00 * p00[1] + w01 * p01[1] + w10 * p10[1] + w11 * p11[1];
}

// ---------------------------------------------------------------------------
// Kernel 3: long refinement — base in registers, 2 samples of L + 2 of S.
// Group index is the FAST thread dim so channel-interleaved stores coalesce.
// ---------------------------------------------------------------------------
__global__ __launch_bounds__(256)
void long_refine(const float* __restrict__ Lfield, const float* __restrict__ Sfield,
                 float* __restrict__ out_long) {
    const long long t = (long long)blockIdx.x * blockDim.x + threadIdx.x;
    if (t >= (long long)17 * NPIX) return;
    const int k   = (int)(t % 17);
    const int pix = (int)(t / 17);
    const int n = pix / HW;
    const int p = pix % HW;
    const int y = p / IMGD, x = p % IMGD;
    const int ch = 2 * k;

    const float* Ln = Lfield + (size_t)n * HW * 34;
    const float* Sn = Sfield + (size_t)n * HW * 34;

    float bx = Ln[(size_t)p * 34 + ch];
    float by = Ln[(size_t)p * 34 + ch + 1];
    float sx, sy;
    sample34(Ln, ch, x, y, bx, by, sx, sy); bx += sx; by += sy;
    sample34(Ln, ch, x, y, bx, by, sx, sy); bx += sx; by += sy;
    sample34(Sn, ch, x, y, bx, by, sx, sy); bx += sx; by += sy;
    sample34(Sn, ch, x, y, bx, by, sx, sy); bx += sx; by += sy;

    out_long[(size_t)pix * 34 + ch]     = bx;
    out_long[(size_t)pix * 34 + ch + 1] = by;
}

// ---------------------------------------------------------------------------
// Kernel 4: mid refinement — base = on-the-fly resize of mo, 2 samples of
// S[to_kp[g]]. Group index fast for coalesced stores.
// ---------------------------------------------------------------------------
__global__ __launch_bounds__(256)
void mid_refine(const float* __restrict__ small, const float* __restrict__ Sfield,
                float* __restrict__ out_mid) {
    const long long t = (long long)blockIdx.x * blockDim.x + threadIdx.x;
    if (t >= (long long)32 * NPIX) return;
    const int g   = (int)(t & 31);
    const int pix = (int)(t >> 5);
    const int n = pix / HW;
    const int p = pix % HW;
    const int y = p / IMGD, x = p % IMGD;

    // on-the-fly resize of mo channels (51 + 2g, 51 + 2g + 1)
    int x0, x1, y0, y1; float wx, wy;
    rs_axis(x, x0, x1, wx);
    rs_axis(y, y0, y1, wy);
    const float w00 = (1.0f - wy) * (1.0f - wx);
    const float w01 = (1.0f - wy) * wx;
    const float w10 = wy * (1.0f - wx);
    const float w11 = wy * wx;
    const int c = 51 + 2 * g;
    const float* s00 = small + (size_t)((n * 13 + y0) * 13 + x0) * CTOT + c;
    const float* s01 = small + (size_t)((n * 13 + y0) * 13 + x1) * CTOT + c;
    const float* s10 = small + (size_t)((n * 13 + y1) * 13 + x0) * CTOT + c;
    const float* s11 = small + (size_t)((n * 13 + y1) * 13 + x1) * CTOT + c;
    float bx = w00 * s00[0] + w01 * s01[0] + w10 * s10[0] + w11 * s11[0];
    float by = w00 * s00[1] + w01 * s01[1] + w10 * s10[1] + w11 * s11[1];

    const float* Sn = Sfield + (size_t)n * HW * 34;
    const int ch = 2 * TO_KP[g];
    float sx, sy;
    sample34(Sn, ch, x, y, bx, by, sx, sy); bx += sx; by += sy;
    sample34(Sn, ch, x, y, bx, by, sx, sy); bx += sx; by += sy;

    out_mid[(size_t)pix * 64 + 2 * g]     = bx;
    out_mid[(size_t)pix * 64 + 2 * g + 1] = by;
}

// ---------------------------------------------------------------------------
extern "C" void kernel_launch(void* const* d_in, const int* in_sizes, int n_in,
                              void* d_out, int out_size, void* d_ws, size_t ws_size,
                              hipStream_t stream) {
    const float* x    = (const float*)d_in[0];
    const float* kp_w = (const float*)d_in[1];
    const float* kp_b = (const float*)d_in[2];
    const float* so_w = (const float*)d_in[3];
    const float* so_b = (const float*)d_in[4];
    const float* mo_w = (const float*)d_in[5];
    const float* mo_b = (const float*)d_in[6];
    const float* lo_w = (const float*)d_in[7];
    const float* lo_b = (const float*)d_in[8];
    const float* ss_w = (const float*)d_in[9];
    const float* ss_b = (const float*)d_in[10];

    float* out = (float*)d_out;
    float* out_kp   = out;                                   // [NPIX][17]
    float* out_so   = out_kp  + (size_t)NPIX * 17;           // [NPIX][34] == S field
    float* out_mid  = out_so  + (size_t)NPIX * 34;           // [NPIX][64]
    float* out_long = out_mid + (size_t)NPIX * 64;           // [NPIX][34]
    float* out_ss   = out_long + (size_t)NPIX * 34;          // [NPIX]

    float* ws      = (float*)d_ws;
    float* small   = ws;                                     // [338][150]
    float* Lfield  = ws + 51200;                             // [NPIX][34] (~43.7 MB)

    // 1) projection GEMM with fused bias/sigmoid (WMMA)
    gemm150_wmma<<<dim3(22, 10), dim3(32), 0, stream>>>(
        x, kp_w, kp_b, so_w, so_b, mo_w, mo_b, lo_w, lo_b, ss_w, ss_b, small);

    // 2) resize: kp/so/ss -> d_out, L -> ws
    {
        int grid = (NPIX + 255) / 256;
        resize_scatter<<<grid, 256, 0, stream>>>(small, out_kp, out_so, out_ss, Lfield);
    }

    // 3) long refinement (reads L in ws + S in d_out)
    {
        long long total = (long long)17 * NPIX;
        int grid = (int)((total + 255) / 256);
        long_refine<<<grid, 256, 0, stream>>>(Lfield, out_so, out_long);
    }

    // 4) mid refinement (on-the-fly mo resize + S samples)
    {
        long long total = (long long)32 * NPIX;
        int grid = (int)((total + 255) / 256);
        mid_refine<<<grid, 256, 0, stream>>>(small, out_so, out_mid);
    }
}
